// CompressedInteractionNetwork_47639777247621
// MI455X (gfx1250) — compile-verified, matrix operand order checked by reading, not run on hardware
//
#include <hip/hip_runtime.h>

// ---------------------------------------------------------------------------
// CIN (Compressed Interaction Network) for MI455X / gfx1250, wave32 + WMMA.
//
//   z[b,o,d] = relu( sum_{m,n} x0[b,m,d]*h[b,n,d]*W[o,m,n] + bias[o] )
//   per batch:  Z(256x64) = W(256xK) @ P(Kx64),  P[(m,n),d] = x0[m,d]*h[n,d]
//
// ~386 GFLOP vs ~22 MB HBM traffic -> compute bound -> f16 WMMA, f32 accum.
// x/h kept TRANSPOSED in LDS (ht[d][n]) so P production is branch-free.
// K processed in 64-wide chunks (double-buffered). Per chunk we use the
// CDNA5 SPLIT barrier: s_wait_dscnt + s_barrier_signal right after the P
// stores, then issue A-tile global loads (they fly while other waves
// arrive), then s_barrier_wait; loadcnt is only waited before the WMMAs.
// W0 is re-packed to a stride-40 virtual K (k'=m*40+n, zero padded to 1600)
// so aligned 8-groups never cross a row in any layer.
// ---------------------------------------------------------------------------

typedef __attribute__((ext_vector_type(16))) _Float16 v16h;
typedef __attribute__((ext_vector_type(8)))  _Float16 v8h;
typedef __attribute__((ext_vector_type(8)))  float    v8f;

#define BATCH 1024
#define MROWS 39
#define DCOLS 64
#define NS0   40        // layer-0 virtual row stride (8 | 40)
#define K0VP  1600      // 39*40 = 1560 padded to multiple of 64 (25 chunks)
#define K12   4992      // 39*128, multiple of 64 (78 chunks)

// ---------------------------------------------------------------------------
// Prepass: repack fp32 weights [256][39*NSRC] -> f16 [256][Kp] with
// k' = m*NS + n; zero where m >= 39 or n >= NSRC.
// ---------------------------------------------------------------------------
__global__ void cin_repack_kernel(const float* __restrict__ src,
                                  _Float16* __restrict__ dst,
                                  int NSRC, int NS, int Kp) {
  int i = blockIdx.x * blockDim.x + threadIdx.x;
  if (i >= 256 * Kp) return;
  int o  = i / Kp;
  int kp = i - o * Kp;
  int m  = kp / NS;
  int n  = kp - m * NS;
  float v = 0.0f;
  if (m < MROWS && n < NSRC)
    v = src[(size_t)o * (MROWS * NSRC) + m * NSRC + n];
  dst[i] = (_Float16)v;
}

__device__ __forceinline__ v8f vzero8() {
  v8f z;
#pragma unroll
  for (int i = 0; i < 8; ++i) z[i] = 0.0f;
  return z;
}

// ---------------------------------------------------------------------------
// One CIN layer. NS = h row stride (40 or 128), KPAD = padded virtual K
// (multiple of 64). 8 waves; wave w owns output rows [32w,32w+32) (two
// 16-row o-tiles) x all four 16-col d-tiles -> 8 f32 WMMA accumulators.
// Each split-barrier covers a 64-wide K chunk: 16 WMMAs per wave.
// ---------------------------------------------------------------------------
template <int NS, int KPAD, bool SPLIT>
__device__ __forceinline__ void cin_layer(
    const _Float16* __restrict__ Wl,   // global f16 [256][KPAD]
    const _Float16* xt,                // LDS x0^T [64][40] (zero padded)
    const _Float16* hsrc,              // LDS h^T  [64][NS]
    _Float16* hdst,                    // LDS next h^T [64][128]
    _Float16* Pbuf,                    // LDS 2 x [64][64] double buffer
    const float* biasl,                // LDS bias [256]
    const float* fcwAll,               // LDS fcW [512]
    float* outacc,                     // LDS scalar accumulator
    int chanBase, int tid) {
  const int lane  = tid & 31;
  const int wid   = tid >> 5;
  const int obase = wid * 32;
  const int dloc  = lane & 15;              // d position inside tile
  const int ksel  = (lane & 16) ? 8 : 0;    // A/B K-half select (ISA layout)
  const int voff  = (lane & 16) ? 8 : 0;    // C/D row offset    (ISA layout)

  v8f acc[2][4];
#pragma unroll
  for (int t = 0; t < 2; ++t)
#pragma unroll
    for (int j = 0; j < 4; ++j) acc[t][j] = vzero8();

  const int myd = tid >> 2;         // 0..63 : d column this thread produces
  const int kq0 = (tid & 3) * 16;   // 0,16,32,48 : k sub-range it fills
  const int nchunk = KPAD / 64;

  for (int kc = 0; kc < nchunk; ++kc) {
    // ---- produce transposed P chunk Pt[d][0..63], branch-free ------------
    _Float16* pt = Pbuf + (kc & 1) * (64 * 64);
#pragma unroll
    for (int g = 0; g < 2; ++g) {
      const int kb = kc * 64 + kq0 + g * 8;   // multiple of 8
      const int m = kb / NS;                  // group-uniform (8 | NS)
      const int n = kb - m * NS;
      const _Float16 xv = xt[myd * NS0 + m];  // padding rows are zero
      const v8h hv = *(const v8h*)(hsrc + myd * NS + n);
      *(v8h*)(pt + myd * 64 + kq0 + g * 8) = hv * xv;  // pk_mul + b128 store
    }

    // ---- split barrier: arrive as soon as P stores are visible -----------
    asm volatile("s_wait_dscnt 0x0\n\ts_barrier_signal -1" ::: "memory");

    // ---- A tiles issued between signal and wait (fly during arrival) -----
    v16h a[2][2];
#pragma unroll
    for (int t = 0; t < 2; ++t) {
      const _Float16* wrow =
          Wl + (size_t)(obase + t * 16 + dloc) * KPAD + kc * 64 + ksel;
      __builtin_prefetch((const void*)(wrow + 128), 0, 1);  // two chunks ahead
#pragma unroll
      for (int s = 0; s < 2; ++s) {
        v8h lo = *(const v8h*)(wrow + s * 32);        // K {0-7} / {8-15}
        v8h hi = *(const v8h*)(wrow + s * 32 + 16);   // K {16-23} / {24-31}
#pragma unroll
        for (int u = 0; u < 8; ++u) { a[t][s][u] = lo[u]; a[t][s][8 + u] = hi[u]; }
      }
    }

    asm volatile("s_barrier_wait -1" ::: "memory");

    // ---- B tiles from LDS P chunk, 16 WMMAs ------------------------------
#pragma unroll
    for (int s = 0; s < 2; ++s) {
#pragma unroll
      for (int j = 0; j < 4; ++j) {
        const _Float16* pb = pt + (dloc + j * 16) * 64 + s * 32 + ksel;
        v8h lo = *(const v8h*)(pb);
        v8h hi = *(const v8h*)(pb + 16);
        v16h bm;
#pragma unroll
        for (int u = 0; u < 8; ++u) { bm[u] = lo[u]; bm[8 + u] = hi[u]; }
        acc[0][j] = __builtin_amdgcn_wmma_f32_16x16x32_f16(
            false, a[0][s], false, bm, (short)0, acc[0][j], false, false);
        acc[1][j] = __builtin_amdgcn_wmma_f32_16x16x32_f16(
            false, a[1][s], false, bm, (short)0, acc[1][j], false, false);
      }
    }
  }

  // ---- epilogue: bias + relu; h half -> LDS (b128); fused fcW reduce -----
  float s = 0.0f;
#pragma unroll
  for (int t = 0; t < 2; ++t) {
    const int r0 = obase + t * 16 + voff;   // row of acc element v=0
#pragma unroll
    for (int j = 0; j < 4; ++j) {
      const int d = dloc + j * 16;
      if (SPLIT && r0 >= 128) {             // wave-uniform branch
        v8h hz;
#pragma unroll
        for (int v = 0; v < 8; ++v) {
          float z = fmaxf(acc[t][j][v] + biasl[r0 + v], 0.0f);
          hz[v] = (_Float16)z;
        }
        *(v8h*)(hdst + d * 128 + (r0 - 128)) = hz;   // contiguous, aligned
      } else {
#pragma unroll
        for (int v = 0; v < 8; ++v) {
          float z = fmaxf(acc[t][j][v] + biasl[r0 + v], 0.0f);
          s += fcwAll[chanBase + r0 + v] * z;
        }
      }
    }
  }
  atomicAdd(outacc, s);   // ds_add_f32 on LDS scalar
  __syncthreads();        // h complete + P buffers free before next layer
}

// ---------------------------------------------------------------------------
// Main kernel: one workgroup (8 waves) per batch element.
// ---------------------------------------------------------------------------
__global__ __launch_bounds__(256) void cin_main_kernel(
    const float* __restrict__ x,
    const float* __restrict__ b0, const float* __restrict__ b1,
    const float* __restrict__ b2,
    const float* __restrict__ fcW, const float* __restrict__ fcb,
    const _Float16* __restrict__ W0p, const _Float16* __restrict__ W1p,
    const _Float16* __restrict__ W2p,
    float* __restrict__ out) {
  __shared__ _Float16 ls_xt[DCOLS * NS0];      //  5.0 KB  x0^T, zero padded
  __shared__ _Float16 ls_ht[DCOLS * 128];      // 16.0 KB  h^T
  __shared__ _Float16 ls_P[2 * 64 * 64];       // 16.0 KB  double buffer
  __shared__ float    ls_fcw[512];             //  2.0 KB
  __shared__ float    ls_bias[3 * 256];        //  3.0 KB
  __shared__ float    ls_out;

  const int tid = threadIdx.x;
  const int b   = blockIdx.x;

  {  // transposed, coalesced x load: xt[d][m] = x[b][m][d]; m=39 row -> 0
    const int d  = tid & 63;
    const int m0 = tid >> 6;                  // 0..3
#pragma unroll
    for (int m = m0; m < NS0; m += 4) {
      float v = (m < MROWS) ? x[(size_t)b * MROWS * DCOLS + m * DCOLS + d]
                            : 0.0f;
      ls_xt[d * NS0 + m] = (_Float16)v;
    }
  }
  for (int i = tid; i < 512; i += 256) ls_fcw[i] = fcW[i];
  if (tid < 256) {
    ls_bias[tid]       = b0[tid];
    ls_bias[256 + tid] = b1[tid];
    ls_bias[512 + tid] = b2[tid];
  }
  if (tid == 0) ls_out = 0.0f;
  __syncthreads();

  // Layer 0: h = x0 (stride 40), K=1600 virt, split -> channels [0,128)
  cin_layer<NS0, K0VP, true>(W0p, ls_xt, ls_xt, ls_ht, ls_P,
                             ls_bias, ls_fcw, &ls_out, 0, tid);
  // Layer 1: NS=128, K=4992, split -> channels [128,256)
  cin_layer<128, K12, true>(W1p, ls_xt, ls_ht, ls_ht, ls_P,
                            ls_bias + 256, ls_fcw, &ls_out, 128, tid);
  // Layer 2: NS=128, K=4992, no split -> channels [256,512)
  cin_layer<128, K12, false>(W2p, ls_xt, ls_ht, ls_ht, ls_P,
                             ls_bias + 512, ls_fcw, &ls_out, 256, tid);

  if (tid == 0) out[b] = ls_out + fcb[0];
}

// ---------------------------------------------------------------------------
extern "C" void kernel_launch(void* const* d_in, const int* in_sizes, int n_in,
                              void* d_out, int out_size, void* d_ws,
                              size_t ws_size, hipStream_t stream) {
  const float* x   = (const float*)d_in[0];
  const float* W0  = (const float*)d_in[1];
  const float* b0  = (const float*)d_in[2];
  const float* W1  = (const float*)d_in[3];
  const float* b1  = (const float*)d_in[4];
  const float* W2  = (const float*)d_in[5];
  const float* b2  = (const float*)d_in[6];
  const float* fcW = (const float*)d_in[7];
  const float* fcb = (const float*)d_in[8];
  float* out = (float*)d_out;

  // f16 repacked weights in workspace: 256*1600 + 2*256*4992 halfs (~5.9 MB)
  _Float16* w0p = (_Float16*)d_ws;
  _Float16* w1p = w0p + (size_t)256 * K0VP;
  _Float16* w2p = w1p + (size_t)256 * K12;

  const int n0  = 256 * K0VP;
  const int n12 = 256 * K12;
  cin_repack_kernel<<<(n0 + 255) / 256, 256, 0, stream>>>(W0, w0p, 39, NS0, K0VP);
  cin_repack_kernel<<<(n12 + 255) / 256, 256, 0, stream>>>(W1, w1p, 128, 128, K12);
  cin_repack_kernel<<<(n12 + 255) / 256, 256, 0, stream>>>(W2, w2p, 128, 128, K12);

  cin_main_kernel<<<BATCH, 256, 0, stream>>>(x, b0, b1, b2, fcW, fcb,
                                             w0p, w1p, w2p, out);
}
